// MoeGate_49048526520562
// MI455X (gfx1250) — matrix operand order repaired and checked by reading, not compiled, not run on hardware
//
#include <hip/hip_runtime.h>
#include <hip/hip_bf16.h>
#include <math.h>

typedef __attribute__((ext_vector_type(16))) __bf16 v16bf;
typedef __attribute__((ext_vector_type(8)))  __bf16 v8bf;
typedef __attribute__((ext_vector_type(8)))  float  v8f;

#define D_FEAT   1024
#define N_EXPERT 64
#define TOK_PER_BLOCK 64
#define KSLAB    128
#define KP       136   // padded bf16 stride per column in LDS slab (bank-conflict avoidance)
#define HP       72    // padded float stride per token row of H in LDS

__device__ __forceinline__ float softplus_f(float z) {
    // stable softplus: log(1+e^z)
    if (z > 20.0f) return z;
    return log1pf(expf(z));
}

__global__ __launch_bounds__(256)
void MoeGate_49048526520562_kernel(const float* __restrict__ x,
                                   const float* __restrict__ noise,
                                   const float* __restrict__ Wg,
                                   const float* __restrict__ Wn,
                                   float* __restrict__ out)
{
    __shared__ __bf16 Wlds[2 * N_EXPERT * KP];      // [mat*64+e][k] bf16, 34816 B
    __shared__ float  Hlds[TOK_PER_BLOCK * HP];     // 18432 B
    __shared__ float  stats[TOK_PER_BLOCK * 4];     // maxv, kth, 1/sum

    const int tid  = threadIdx.x;
    const int wave = tid >> 5;
    const int lane = tid & 31;
    const int m    = lane & 15;   // row within 16-tile (A) / col within 16-tile (B)
    const int half = lane >> 4;

    const int blockTok = blockIdx.x * TOK_PER_BLOCK;
    const int tt  = wave & 3;                 // token tile 0..3
    const int eh  = wave >> 2;                // expert half 0..1
    const int tokBase = blockTok + tt * 16;
    const int et0 = eh * 32;                  // expert tiles et0, et0+16

    v8f accG0 = {}, accG1 = {}, accN0 = {}, accN1 = {};

    const float* __restrict__ xrow = x + (size_t)(tokBase + m) * D_FEAT;

    for (int kc = 0; kc < D_FEAT; kc += KSLAB) {
        // ---- stage W slab (both matrices) into LDS as bf16, transposed [col][k] ----
        __syncthreads();
        #pragma unroll
        for (int i = 0; i < 64; ++i) {
            int flat = tid + i * 256;           // 0..16383 = 2 mats * 128 k * 64 e
            int mat  = flat >> 13;              // uniform per i
            int rem  = flat & 8191;
            int k    = rem >> 6;
            int e    = rem & 63;
            const float* Wsrc = mat ? Wn : Wg;
            float wv = Wsrc[(size_t)(kc + k) * N_EXPERT + e];
            Wlds[(mat * N_EXPERT + e) * KP + k] = (__bf16)wv;
        }
        __syncthreads();

        // prefetch next x slab for this lane's row
        if (kc + KSLAB < D_FEAT) {
            __builtin_prefetch(xrow + kc + KSLAB + half * 64, 0, 0);
        }

        // ---- inner K loop: 4 steps of K=32 ----
        #pragma unroll
        for (int ks = 0; ks < KSLAB; ks += 32) {
            // A fragment: 16x32 bf16.  lanes 0-15: K {0..7,16..23}; lanes 16-31: K {8..15,24..31}
            const float* ap = xrow + kc + ks + half * 8;
            float4 a0 = *(const float4*)(ap +  0);
            float4 a1 = *(const float4*)(ap +  4);
            float4 a2 = *(const float4*)(ap + 16);
            float4 a3 = *(const float4*)(ap + 20);
            v16bf a;
            a[ 0]=(__bf16)a0.x; a[ 1]=(__bf16)a0.y; a[ 2]=(__bf16)a0.z; a[ 3]=(__bf16)a0.w;
            a[ 4]=(__bf16)a1.x; a[ 5]=(__bf16)a1.y; a[ 6]=(__bf16)a1.z; a[ 7]=(__bf16)a1.w;
            a[ 8]=(__bf16)a2.x; a[ 9]=(__bf16)a2.y; a[10]=(__bf16)a2.z; a[11]=(__bf16)a2.w;
            a[12]=(__bf16)a3.x; a[13]=(__bf16)a3.y; a[14]=(__bf16)a3.z; a[15]=(__bf16)a3.w;

            // B fragments: 32x16 bf16. lane n = col, lanes 0-15: K {0..7,16..23}, lanes 16-31: K {8..15,24..31}
            auto loadB = [&](int col) -> v16bf {
                const __bf16* p = &Wlds[col * KP + ks + half * 8];
                v8bf lo = *(const v8bf*)(p + 0);   // ds_load_b128
                v8bf hi = *(const v8bf*)(p + 16);  // ds_load_b128
                v16bf r;
                #pragma unroll
                for (int j = 0; j < 8; ++j) { r[j] = lo[j]; r[8 + j] = hi[j]; }
                return r;
            };
            v16bf bg0 = loadB(      et0      + m);
            v16bf bg1 = loadB(      et0 + 16 + m);
            v16bf bn0 = loadB(64 +  et0      + m);
            v16bf bn1 = loadB(64 +  et0 + 16 + m);

            accG0 = __builtin_amdgcn_wmma_f32_16x16x32_bf16(false, a, false, bg0, (short)0, accG0, false, false);
            accG1 = __builtin_amdgcn_wmma_f32_16x16x32_bf16(false, a, false, bg1, (short)0, accG1, false, false);
            accN0 = __builtin_amdgcn_wmma_f32_16x16x32_bf16(false, a, false, bn0, (short)0, accN0, false, false);
            accN1 = __builtin_amdgcn_wmma_f32_16x16x32_bf16(false, a, false, bn1, (short)0, accN1, false, false);
        }
    }

    // ---- epilogue: H = Hg + noise * softplus(Hn), spill to LDS ----
    // C/D layout: VGPR r, lanes 0-15 -> M=r, lanes 16-31 -> M=r+8; N = lane&15
    {
        const int n = lane & 15;
        #pragma unroll
        for (int r = 0; r < 8; ++r) {
            int row = r + half * 8;
            int tok = tokBase + row;
            const float* nz = noise + (size_t)tok * N_EXPERT;
            float h0 = accG0[r] + nz[et0 +      n] * softplus_f(accN0[r]);
            float h1 = accG1[r] + nz[et0 + 16 + n] * softplus_f(accN1[r]);
            Hlds[(tt * 16 + row) * HP + et0 +      n] = h0;
            Hlds[(tt * 16 + row) * HP + et0 + 16 + n] = h1;
        }
    }
    __syncthreads();

    // ---- phase 2: one thread per token, find kth (8th largest, tie-correct) ----
    if (tid < TOK_PER_BLOCK) {
        const float* h = &Hlds[tid * HP];
        float maxv = -INFINITY;
        #pragma unroll 8
        for (int e = 0; e < N_EXPERT; ++e) maxv = fmaxf(maxv, h[e]);

        float cur = INFINITY;   // exclusive upper bound for next distinct value
        float kth = maxv;
        int   rem = 8;
        for (int pass = 0; pass < 8 && rem > 0; ++pass) {
            float mx = -INFINITY; int cnt = 0;
            for (int e = 0; e < N_EXPERT; ++e) {
                float v = h[e];
                if (v < cur) {
                    if (v > mx)       { mx = v; cnt = 1; }
                    else if (v == mx) { cnt++; }
                }
            }
            if (cnt == 0) break;
            kth = mx; cur = mx; rem -= cnt;
        }

        float sum = 0.0f;
        for (int e = 0; e < N_EXPERT; ++e) {
            float v = h[e];
            if (v >= kth) sum += expf(v - maxv);
        }
        stats[tid * 4 + 0] = maxv;
        stats[tid * 4 + 1] = kth;
        stats[tid * 4 + 2] = 1.0f / sum;
    }
    __syncthreads();

    // ---- phase 3: coalesced write of G ----
    #pragma unroll
    for (int i = 0; i < 16; ++i) {
        int flat = tid + i * 256;        // 0..4095 = 64 tok * 64 exp
        int t = flat >> 6;
        int e = flat & 63;
        float v    = Hlds[t * HP + e];
        float maxv = stats[t * 4 + 0];
        float kth  = stats[t * 4 + 1];
        float rsum = stats[t * 4 + 2];
        float g = (v >= kth) ? expf(v - maxv) * rsum : 0.0f;
        out[(size_t)(blockTok + t) * N_EXPERT + e] = g;
    }
}

extern "C" void kernel_launch(void* const* d_in, const int* in_sizes, int n_in,
                              void* d_out, int out_size, void* d_ws, size_t ws_size,
                              hipStream_t stream) {
    const float* x     = (const float*)d_in[0];
    const float* noise = (const float*)d_in[1];
    const float* Wg    = (const float*)d_in[2];
    const float* Wn    = (const float*)d_in[3];
    float* out = (float*)d_out;

    int n_tokens = in_sizes[0] / D_FEAT;          // 262144
    int grid = n_tokens / TOK_PER_BLOCK;          // 4096
    MoeGate_49048526520562_kernel<<<grid, 256, 0, stream>>>(x, noise, Wg, Wn, out);
}
